// E3EquivariantBlock_10720238370922
// MI455X (gfx1250) — compile-verified, matrix operand order checked by reading, not run on hardware
//
#include <hip/hip_runtime.h>
#include <math.h>

// ---------------------------------------------------------------------------
// MI455X (gfx1250) fused E(3)-equivariant GNN block.
// wave32, WMMA 16x16x32 f16->f32. Weights pre-packed to f16 fragments (L2
// resident: ~0.5MB), per-edge-tile activations staged in LDS (f16, swizzled
// directly into WMMA A-fragment layout), fp32 atomics for segment_sum.
// TILE_E=32: each wave owns one 16-col N-tile and two 16-edge M-tiles, so
// every B fragment fetched from L2 feeds two v_wmma ops.
// ---------------------------------------------------------------------------

typedef __attribute__((ext_vector_type(16))) _Float16 v16h;
typedef __attribute__((ext_vector_type(8)))  float    v8f;

#define N_NODES 25000
#define N_EDGES 400000
#define H 128
#define D 64
#define MDIM 192        // H + D
#define NLAYERS 3
#define TILE_E 32       // two 16-row M-tiles per block

__device__ __forceinline__ float fastrcp(float x){ return __builtin_amdgcn_rcpf(x); }
__device__ __forceinline__ float siluf(float x){ return x * fastrcp(1.f + __expf(-x)); }
__device__ __forceinline__ float sigmf_(float x){ return fastrcp(1.f + __expf(-x)); }

__device__ __forceinline__ float wave_sum(float v){
  v += __shfl_xor(v, 1, 32);
  v += __shfl_xor(v, 2, 32);
  v += __shfl_xor(v, 4, 32);
  v += __shfl_xor(v, 8, 32);
  v += __shfl_xor(v, 16, 32);
  return v;
}
// reduce across the 16 lanes of one EXEC half (xor masks < 16 stay in half)
__device__ __forceinline__ float half16_sum(float v){
  v += __shfl_xor(v, 1, 32);
  v += __shfl_xor(v, 2, 32);
  v += __shfl_xor(v, 4, 32);
  v += __shfl_xor(v, 8, 32);
  return v;
}

__device__ __forceinline__ void atomAddF(float* p, float v){
  __hip_atomic_fetch_add(p, v, __ATOMIC_RELAXED, __HIP_MEMORY_SCOPE_AGENT);
}

// A-fragment swizzle for V_WMMA_F32_16X16X32_F16 (16-bit A, 16x32):
// lane = h*16 + m ; halves i=0..7 -> K = h*8+i ; i=8..15 -> K = 16+h*8+(i-8)
// storage (two M-tiles): tile[kt*1024 + mt*512 + lane*16 + i]
__device__ __forceinline__ int aswz(int m, int k){
  int kt = k >> 5, kk = k & 31;
  int h  = (kk >> 3) & 1;
  int i  = (kk & 7) + ((kk & 16) ? 8 : 0);
  return kt*1024 + (m >> 4)*512 + (h*16 + (m & 15))*16 + i;
}

// Pack a fp32 weight matrix [K][128] into f16 B-fragments:
// out[((kt*128 + col)*2 + h)*16 + i] = w[(32*kt + kidx(h,i))*128 + col]
__global__ void pack_w_f16(const float* __restrict__ w, _Float16* __restrict__ out, int K){
  int total = K * 128;
  for (int t = blockIdx.x*blockDim.x + threadIdx.x; t < total; t += gridDim.x*blockDim.x){
    int kt = t >> 12;         // / (128*32)
    int r  = t & 4095;
    int c  = r >> 5;
    int kk = r & 31;
    int h  = (kk >> 3) & 1;
    int i  = (kk & 7) + ((kk & 16) ? 8 : 0);
    out[(((kt*128 + c)*2 + h) << 4) + i] = (_Float16)w[(size_t)((kt<<5) + kk)*128 + c];
  }
}

// two M-tiles against one shared B fragment per k-step
__device__ __forceinline__ void gemm_tile2(const _Float16* Atile, const _Float16* __restrict__ Bpack,
                                           int nkt, int col, int h, v8f& acc0, v8f& acc1){
  const int lane = threadIdx.x & 31;
  for (int kt = 0; kt < nkt; ++kt){
    v16h bf = *(const v16h*)(Bpack + (((kt*128 + col)*2 + h) << 4));
    v16h a0 = *(const v16h*)(Atile + ((kt*64      + lane) << 4));
    v16h a1 = *(const v16h*)(Atile + ((kt*64 + 32 + lane) << 4));
    acc0 = __builtin_amdgcn_wmma_f32_16x16x32_f16(false, a0, false, bf, (short)0, acc0, false, false);
    acc1 = __builtin_amdgcn_wmma_f32_16x16x32_f16(false, a1, false, bf, (short)0, acc1, false, false);
  }
}

__global__ void copy_f32(const float* __restrict__ a, float* __restrict__ b, int n){
  int i = blockIdx.x*blockDim.x + threadIdx.x;
  if (i < n) b[i] = a[i];
}

// -------- per-layer node LayerNorm (+ zero aggregators) -------------------
__global__ __launch_bounds__(128)
void node_ln_kernel(const float* __restrict__ x, const float* __restrict__ pos,
                    const float* __restrict__ g, const float* __restrict__ b,
                    const float* __restrict__ pg, const float* __restrict__ pb,
                    float* __restrict__ xn, float* __restrict__ pn,
                    float* __restrict__ s_aggr, float* __restrict__ v_aggr){
  int n = blockIdx.x, t = threadIdx.x;
  __shared__ float rs[4], rq[4];
  float v = x[(size_t)n*H + t];
  float s = wave_sum(v), q = wave_sum(v*v);
  if ((t & 31) == 0){ rs[t>>5] = s; rq[t>>5] = q; }
  __syncthreads();
  float S = rs[0]+rs[1]+rs[2]+rs[3];
  float Q = rq[0]+rq[1]+rq[2]+rq[3];
  float mean = S * (1.f/H);
  float var  = Q * (1.f/H) - mean*mean;
  float ri   = rsqrtf(var + 1e-6f);
  xn[(size_t)n*H + t] = (v - mean)*ri*g[t] + b[t];
  s_aggr[(size_t)n*H + t] = 0.f;
  if (t == 0){
    float p0 = pos[n*3], p1 = pos[n*3+1], p2 = pos[n*3+2];
    float pm = (p0+p1+p2)*(1.f/3.f);
    float pv = ((p0-pm)*(p0-pm)+(p1-pm)*(p1-pm)+(p2-pm)*(p2-pm))*(1.f/3.f);
    float pri = rsqrtf(pv + 1e-6f);
    pn[n*3]   = (p0-pm)*pri*pg[0] + pb[0];
    pn[n*3+1] = (p1-pm)*pri*pg[1] + pb[1];
    pn[n*3+2] = (p2-pm)*pri*pg[2] + pb[2];
    v_aggr[n*3] = 0.f; v_aggr[n*3+1] = 0.f; v_aggr[n*3+2] = 0.f;
  }
}

// -------- fused per-edge MLPs: 4 WMMA GEMMs + LN + atomic scatter ---------
__global__ __launch_bounds__(256)
void edge_kernel(const float* __restrict__ xn, const float* __restrict__ pn,
                 const int* __restrict__ ei, const float* __restrict__ edge_attr,
                 const _Float16* __restrict__ pwa, const _Float16* __restrict__ pws1,
                 const _Float16* __restrict__ pwv, const _Float16* __restrict__ pws2,
                 const float* __restrict__ s_ln1_g, const float* __restrict__ s_ln1_b,
                 const float* __restrict__ v_ln_g, const float* __restrict__ v_ln_b,
                 const float* __restrict__ s_b1, const float* __restrict__ s_b2,
                 const float* __restrict__ s_ln2_g, const float* __restrict__ s_ln2_b,
                 const float* __restrict__ a_b1, const float* __restrict__ a_w2, const float* __restrict__ a_b2,
                 const float* __restrict__ v_b1, const float* __restrict__ v_w2, const float* __restrict__ v_b2,
                 float* __restrict__ s_aggr, float* __restrict__ v_aggr){
  const int tid  = threadIdx.x;
  const int lane = tid & 31;
  const int wv   = tid >> 5;           // 8 waves = 8 N-tiles of 16 columns
  const int eb   = blockIdx.x * TILE_E;

  __shared__ __attribute__((aligned(32))) _Float16 shA0[6*1024];  // raw msg (attn)
  __shared__ __attribute__((aligned(32))) _Float16 shA1[6*1024];  // s_ln1(msg)
  __shared__ __attribute__((aligned(32))) _Float16 shA2[6*1024];  // v_ln(msg)
  __shared__ __attribute__((aligned(32))) _Float16 shH [4*1024];  // silu hidden for s_w2
  __shared__ float sh_mean[TILE_E], sh_rinv[TILE_E];
  __shared__ float sh_red[8][TILE_E], sh_red2[8][TILE_E];
  __shared__ float sh_attn[TILE_E], sh_vw[TILE_E];
  __shared__ float sh_reln[TILE_E][3];
  __shared__ float sh_m2[TILE_E], sh_r2[TILE_E];
  __shared__ int   sh_src[TILE_E], sh_dst[TILE_E];

  if (tid < TILE_E){
    int e = eb + tid; if (e >= N_EDGES) e = N_EDGES - 1;
    int s = ei[e], d = ei[N_EDGES + e];
    sh_src[tid] = s; sh_dst[tid] = d;
    float rx = pn[s*3]   - pn[d*3];
    float ry = pn[s*3+1] - pn[d*3+1];
    float rz = pn[s*3+2] - pn[d*3+2];
    float dist = fmaxf(sqrtf(rx*rx + ry*ry + rz*rz), 1e-6f);
    float rd = fastrcp(dist);
    sh_reln[tid][0] = rx*rd; sh_reln[tid][1] = ry*rd; sh_reln[tid][2] = rz*rd;
  }
  __syncthreads();

  // gather msg = [xn[src] , edge_attr] -> raw f16 A-tile (swizzled)
  for (int idx = tid; idx < TILE_E*MDIM; idx += 256){
    int e = idx / MDIM, k = idx - e*MDIM;
    int ee = eb + e; if (ee >= N_EDGES) ee = N_EDGES - 1;
    float v = (k < H) ? xn[(size_t)sh_src[e]*H + k]
                      : edge_attr[(size_t)ee*D + (k - H)];
    shA0[aswz(e, k)] = (_Float16)v;
  }
  __syncthreads();

  // per-edge LN stats over 192 elements (one wave per 4 edges)
  for (int rep = 0; rep < 4; ++rep){
    int e = wv + rep*8;
    float s = 0.f, q = 0.f;
    for (int k = lane; k < MDIM; k += 32){
      float v = (float)shA0[aswz(e, k)];
      s += v; q += v*v;
    }
    s = wave_sum(s); q = wave_sum(q);
    if (lane == 0){
      float mn = s*(1.f/MDIM);
      float vr = q*(1.f/MDIM) - mn*mn;
      sh_mean[e] = mn; sh_rinv[e] = rsqrtf(vr + 1e-6f);
    }
  }
  __syncthreads();

  // build the two LayerNorm'd f16 A-tiles from the raw tile
  for (int idx = tid; idx < TILE_E*MDIM; idx += 256){
    int e = idx / MDIM, k = idx - e*MDIM;
    int p = aswz(e, k);
    float v  = (float)shA0[p];
    float nm = (v - sh_mean[e]) * sh_rinv[e];
    shA1[p] = (_Float16)(nm * s_ln1_g[k] + s_ln1_b[k]);
    shA2[p] = (_Float16)(nm * v_ln_g[k] + v_ln_b[k]);
  }
  __syncthreads();

  const int col = wv*16 + (lane & 15);
  const int hh  = lane >> 4;

  // ---- attention branch: sigmoid( silu(msg@a_w1+a_b1) @ a_w2 + a_b2 ) ----
  {
    v8f acc0 = {}, acc1 = {};
    gemm_tile2(shA0, pwa, 6, col, hh, acc0, acc1);
    float b1c = a_b1[col], w2c = a_w2[col];
    #pragma unroll
    for (int j = 0; j < 8; ++j){
      float h0 = half16_sum(siluf(acc0[j] + b1c) * w2c);
      float h1 = half16_sum(siluf(acc1[j] + b1c) * w2c);
      if ((lane & 15) == 0){
        sh_red[wv][hh*8 + j]      = h0;
        sh_red[wv][16 + hh*8 + j] = h1;
      }
    }
  }
  __syncthreads();
  if (tid < TILE_E){
    float s = a_b2[0];
    for (int w = 0; w < 8; ++w) s += sh_red[w][tid];
    sh_attn[tid] = sigmf_(s);
  }
  __syncthreads();

  // ---- vector branch: silu(ln_v(msg)@v_w1+v_b1) @ v_w2 + v_b2 -----------
  {
    v8f acc0 = {}, acc1 = {};
    gemm_tile2(shA2, pwv, 6, col, hh, acc0, acc1);
    float b1c = v_b1[col], w2c = v_w2[col];
    #pragma unroll
    for (int j = 0; j < 8; ++j){
      float h0 = half16_sum(siluf(acc0[j] + b1c) * w2c);
      float h1 = half16_sum(siluf(acc1[j] + b1c) * w2c);
      if ((lane & 15) == 0){
        sh_red[wv][hh*8 + j]      = h0;
        sh_red[wv][16 + hh*8 + j] = h1;
      }
    }
  }
  __syncthreads();
  if (tid < TILE_E){
    float s = v_b2[0];
    for (int w = 0; w < 8; ++w) s += sh_red[w][tid];
    sh_vw[tid] = s;
  }
  __syncthreads();

  // ---- scalar branch stage 1: silu(ln_s1(msg)@s_w1 + s_b1) -> f16 tile ---
  {
    v8f acc0 = {}, acc1 = {};
    gemm_tile2(shA1, pws1, 6, col, hh, acc0, acc1);
    float b1c = s_b1[col];
    #pragma unroll
    for (int j = 0; j < 8; ++j){
      shH[aswz(hh*8 + j,      col)] = (_Float16)siluf(acc0[j] + b1c);
      shH[aswz(16 + hh*8 + j, col)] = (_Float16)siluf(acc1[j] + b1c);
    }
  }
  __syncthreads();

  // ---- stage 2: hidden @ s_w2 + s_b2, LN(s_ln2) * attn, atomic scatter ---
  {
    v8f acc0 = {}, acc1 = {};
    gemm_tile2(shH, pws2, 4, col, hh, acc0, acc1);
    float b2c = s_b2[col];
    float h2v[2][8];
    #pragma unroll
    for (int j = 0; j < 8; ++j){ h2v[0][j] = acc0[j] + b2c; h2v[1][j] = acc1[j] + b2c; }
    #pragma unroll
    for (int mt = 0; mt < 2; ++mt){
      #pragma unroll
      for (int j = 0; j < 8; ++j){
        float s = half16_sum(h2v[mt][j]);
        float q = half16_sum(h2v[mt][j]*h2v[mt][j]);
        if ((lane & 15) == 0){
          sh_red [wv][mt*16 + hh*8 + j] = s;
          sh_red2[wv][mt*16 + hh*8 + j] = q;
        }
      }
    }
    __syncthreads();
    if (tid < TILE_E){
      float S = 0.f, Q = 0.f;
      for (int w = 0; w < 8; ++w){ S += sh_red[w][tid]; Q += sh_red2[w][tid]; }
      float mn = S*(1.f/H);
      float vr = Q*(1.f/H) - mn*mn;
      sh_m2[tid] = mn; sh_r2[tid] = rsqrtf(vr + 1e-6f);
    }
    __syncthreads();
    float gC = s_ln2_g[col], bC = s_ln2_b[col];
    #pragma unroll
    for (int mt = 0; mt < 2; ++mt){
      #pragma unroll
      for (int j = 0; j < 8; ++j){
        int m = mt*16 + hh*8 + j;
        if (eb + m < N_EDGES){
          float val = ((h2v[mt][j] - sh_m2[m])*sh_r2[m]*gC + bC) * sh_attn[m];
          atomAddF(&s_aggr[(size_t)sh_dst[m]*H + col], val);
        }
      }
    }
  }

  if (tid < TILE_E*3){
    int e = tid/3, k = tid - e*3;
    if (eb + e < N_EDGES){
      float val = sh_vw[e] * sh_reln[e][k] * sh_attn[e];
      atomAddF(&v_aggr[sh_dst[e]*3 + k], val);
    }
  }
}

// -------- node gate update ------------------------------------------------
__global__ __launch_bounds__(128)
void node_update_kernel(const float* __restrict__ xn, const float* __restrict__ pn,
                        const float* __restrict__ s_aggr, const float* __restrict__ v_aggr,
                        const float* __restrict__ g_w, const float* __restrict__ g_b,
                        float* __restrict__ xcur, float* __restrict__ pcur){
  int n = blockIdx.x, t = threadIdx.x;
  __shared__ float cat[2*H];
  cat[t]     = xn[(size_t)n*H + t];
  cat[H + t] = s_aggr[(size_t)n*H + t];
  __syncthreads();
  float a = g_b[t];
  #pragma unroll 8
  for (int k = 0; k < 2*H; ++k) a += cat[k] * g_w[(size_t)k*H + t];
  float gate = sigmf_(a);
  xcur[(size_t)n*H + t] = cat[t]*(1.f - gate) + cat[H + t]*gate;
  if (t < 3){
    float p = pn[n*3 + t] + v_aggr[n*3 + t];
    pcur[n*3 + t] = fminf(fmaxf(p, -10.f), 10.f);
  }
}

// -------- energy head + output assembly ----------------------------------
__global__ __launch_bounds__(128)
void energy_kernel(const float* __restrict__ xcur, const float* __restrict__ pcur,
                   const float* __restrict__ e_w1, const float* __restrict__ e_b1,
                   const float* __restrict__ e_w2, const float* __restrict__ e_b2,
                   float* __restrict__ out){
  int n = blockIdx.x, t = threadIdx.x;
  __shared__ float xr[H];
  __shared__ float h1[H];
  xr[t] = xcur[(size_t)n*H + t];
  __syncthreads();
  float a = e_b1[t];
  #pragma unroll 8
  for (int k = 0; k < H; ++k) a += xr[k] * e_w1[(size_t)k*H + t];
  h1[t] = fmaxf(a, 0.f);
  __syncthreads();
  float o = e_b2[t];
  #pragma unroll 8
  for (int k = 0; k < H; ++k) o += h1[k] * e_w2[(size_t)k*H + t];
  out[(size_t)n*H + t] = o;
  if (t < 3) out[(size_t)N_NODES*H + n*3 + t] = pcur[n*3 + t];
}

extern "C" void kernel_launch(void* const* d_in, const int* in_sizes, int n_in,
                              void* d_out, int out_size, void* d_ws, size_t ws_size,
                              hipStream_t stream){
  const float* x         = (const float*)d_in[0];
  const float* pos       = (const float*)d_in[1];
  const int*   ei        = (const int*)  d_in[2];
  const float* edge_attr = (const float*)d_in[3];
  const float* ln_x_g  = (const float*)d_in[4];
  const float* ln_x_b  = (const float*)d_in[5];
  const float* ln_p_g  = (const float*)d_in[6];
  const float* ln_p_b  = (const float*)d_in[7];
  const float* s_ln1_g = (const float*)d_in[8];
  const float* s_ln1_b = (const float*)d_in[9];
  const float* s_w1    = (const float*)d_in[10];
  const float* s_b1    = (const float*)d_in[11];
  const float* s_w2    = (const float*)d_in[12];
  const float* s_b2    = (const float*)d_in[13];
  const float* s_ln2_g = (const float*)d_in[14];
  const float* s_ln2_b = (const float*)d_in[15];
  const float* v_ln_g  = (const float*)d_in[16];
  const float* v_ln_b  = (const float*)d_in[17];
  const float* v_w1    = (const float*)d_in[18];
  const float* v_b1    = (const float*)d_in[19];
  const float* v_w2    = (const float*)d_in[20];
  const float* v_b2    = (const float*)d_in[21];
  const float* a_w1    = (const float*)d_in[22];
  const float* a_b1    = (const float*)d_in[23];
  const float* a_w2    = (const float*)d_in[24];
  const float* a_b2    = (const float*)d_in[25];
  const float* g_w     = (const float*)d_in[26];
  const float* g_b     = (const float*)d_in[27];
  const float* e_w1    = (const float*)d_in[28];
  const float* e_b1    = (const float*)d_in[29];
  const float* e_w2    = (const float*)d_in[30];
  const float* e_b2    = (const float*)d_in[31];

  // workspace layout (fp32 region then f16 packed-weight region, 32B aligned)
  float* ws   = (float*)d_ws;
  float* xcur = ws;
  float* xn   = xcur + (size_t)N_NODES*H;
  float* s_ag = xn   + (size_t)N_NODES*H;
  float* pcur = s_ag + (size_t)N_NODES*H;
  float* pn   = pcur + (size_t)N_NODES*3;
  float* v_ag = pn   + (size_t)N_NODES*3;
  _Float16* pw = (_Float16*)(v_ag + (size_t)N_NODES*3);
  const size_t A1    = (size_t)MDIM*H;          // 24576 halves per [192,128] matrix
  const size_t PER_L = 3*A1 + (size_t)H*H;      // + 16384 for s_w2

  copy_f32<<<(N_NODES*H + 255)/256, 256, 0, stream>>>(x, xcur, N_NODES*H);
  copy_f32<<<(N_NODES*3 + 255)/256, 256, 0, stream>>>(pos, pcur, N_NODES*3);

  for (int l = 0; l < NLAYERS; ++l){
    pack_w_f16<<<96, 256, 0, stream>>>(a_w1 + (size_t)l*MDIM*H, pw + l*PER_L + 0*A1, MDIM);
    pack_w_f16<<<96, 256, 0, stream>>>(s_w1 + (size_t)l*MDIM*H, pw + l*PER_L + 1*A1, MDIM);
    pack_w_f16<<<96, 256, 0, stream>>>(v_w1 + (size_t)l*MDIM*H, pw + l*PER_L + 2*A1, MDIM);
    pack_w_f16<<<64, 256, 0, stream>>>(s_w2 + (size_t)l*H*H,    pw + l*PER_L + 3*A1, H);
  }

  for (int l = 0; l < NLAYERS; ++l){
    node_ln_kernel<<<N_NODES, 128, 0, stream>>>(
        xcur, pcur, ln_x_g + l*H, ln_x_b + l*H, ln_p_g + l*3, ln_p_b + l*3,
        xn, pn, s_ag, v_ag);

    edge_kernel<<<(N_EDGES + TILE_E - 1)/TILE_E, 256, 0, stream>>>(
        xn, pn, ei, edge_attr,
        pw + l*PER_L + 0*A1, pw + l*PER_L + 1*A1, pw + l*PER_L + 2*A1, pw + l*PER_L + 3*A1,
        s_ln1_g + (size_t)l*MDIM, s_ln1_b + (size_t)l*MDIM,
        v_ln_g  + (size_t)l*MDIM, v_ln_b  + (size_t)l*MDIM,
        s_b1 + l*H, s_b2 + l*H, s_ln2_g + l*H, s_ln2_b + l*H,
        a_b1 + l*H, a_w2 + l*H, a_b2 + l,
        v_b1 + l*H, v_w2 + l*H, v_b2 + l,
        s_ag, v_ag);

    node_update_kernel<<<N_NODES, 128, 0, stream>>>(
        xn, pn, s_ag, v_ag, g_w + (size_t)l*2*H*H, g_b + l*H, xcur, pcur);
  }

  energy_kernel<<<N_NODES, 128, 0, stream>>>(
      xcur, pcur, e_w1, e_b1, e_w2, e_b2, (float*)d_out);
}
